// LSTMLayer_34462817583252
// MI455X (gfx1250) — matrix operand order repaired and checked by reading, not compiled, not run on hardware
//
#include <hip/hip_runtime.h>
#include <hip/hip_bf16.h>

// LSTM w/ layernorm: B=64, T=1024, I=O=512, 4*O=2048 gate outputs.
// Plan:
//  1) convert W -> Wxb(2048x512 bf16), Whb(2048x512 bf16); convert x -> bf16
//  2) one big WMMA bf16 GEMM: Zx[t][b][n] = x @ Wx^T  (no recurrence)
//  3) persistent recurrent kernel: 4 WGs x 1024 thr; WG g owns batch rows
//     [16g,16g+16). Per step: Zh = h @ Wh^T via v_wmma_f32_16x16x32_bf16
//     (A from LDS, B from L2-resident Whb), + Zx tile, layernorm stats via
//     LDS atomics, gates/gelu, h/c update all inside the WGP (s_barrier only).
// Workspace: Wxb 2MB | Whb 2MB | xb 64MB | Zx 512MB  (~581 MB total)

typedef __attribute__((ext_vector_type(16))) __bf16        v16bf;
typedef __attribute__((ext_vector_type(8)))  float         v8f;
typedef __attribute__((ext_vector_type(8)))  unsigned short v8us;

union BFrag { v16bf v; struct { v8us lo, hi; } h; };

__device__ __forceinline__ unsigned short f2bf(float f) {
  unsigned u = __float_as_uint(f);
  unsigned rb = ((u >> 16) & 1u) + 0x7fffu;   // round-to-nearest-even
  return (unsigned short)((u + rb) >> 16);
}

// ---------------- conversion kernels ----------------
__global__ void k_conv_w(const float* __restrict__ W,
                         unsigned short* __restrict__ Wxb,
                         unsigned short* __restrict__ Whb) {
  int idx = blockIdx.x * 256 + threadIdx.x;      // 0 .. 2048*1024-1
  int n = idx >> 10, k = idx & 1023;
  unsigned short v = f2bf(W[idx]);
  if (k < 512) Wxb[n * 512 + k] = v;
  else         Whb[n * 512 + (k - 512)] = v;
}

__global__ void k_conv_x(const float* __restrict__ x,
                         unsigned short* __restrict__ xb) {
  size_t idx = (size_t)blockIdx.x * 256 + threadIdx.x;  // B*T*I elements
  xb[idx] = f2bf(x[idx]);
}

// ---------------- phase 1: Zx = x @ Wx^T ----------------
// M = B*T = 65536 (row r = b*1024 + t), N = 2048, K = 512.
// One 16x16 output tile per wave; 8 waves/block; 65536 blocks.
__global__ void k_gemm_zx(const unsigned short* __restrict__ xb,
                          const unsigned short* __restrict__ Wxb,
                          float* __restrict__ Zx) {
  const int lane = threadIdx.x & 31;
  const int gw   = blockIdx.x * 8 + (threadIdx.x >> 5);
  const int mt   = gw >> 7;            // 4096 M-tiles
  const int nt   = gw & 127;           // 128  N-tiles
  const int r0   = mt * 16;
  const int mA   = lane & 15;
  const int asel = (lane < 16) ? 0 : 8;    // A: K half-chunk select
  const int bsel = (lane < 16) ? 0 : 8;    // B: K half-chunk select
  const unsigned short* arow = xb  + (size_t)(r0 + mA) * 512;
  const unsigned short* brow = Wxb + (size_t)(nt * 16 + (lane & 15)) * 512;

  v8f acc = {0.f, 0.f, 0.f, 0.f, 0.f, 0.f, 0.f, 0.f};
  for (int k0 = 0; k0 < 512; k0 += 32) {
    BFrag a, b;
    a.h.lo = *(const v8us*)(arow + k0 + asel);
    a.h.hi = *(const v8us*)(arow + k0 + 16 + asel);
    b.h.lo = *(const v8us*)(brow + k0 + bsel);
    b.h.hi = *(const v8us*)(brow + k0 + 16 + bsel);
    acc = __builtin_amdgcn_wmma_f32_16x16x32_bf16(
        false, a.v, false, b.v, (short)0, acc, false, false);
  }
#pragma unroll
  for (int r = 0; r < 8; ++r) {
    int m  = (lane < 16) ? r : (8 + r);
    int rg = r0 + m;
    int bb = rg >> 10, tt = rg & 1023;
    Zx[((size_t)tt * 64 + bb) * 2048 + nt * 16 + (lane & 15)] = acc[r];
  }
}

// ---------------- phase 2: recurrent loop ----------------
__global__ void __launch_bounds__(1024, 1)
k_lstm(const float* __restrict__ Zx, const unsigned short* __restrict__ Whb,
       const float* __restrict__ gamma, const float* __restrict__ beta,
       const float* __restrict__ h0, const float* __restrict__ c0,
       float* __restrict__ out) {
  extern __shared__ char smem[];
  float* z_lds = (float*)smem;                          // 16*2048 f32 (128 KB)
  float* c_lds = z_lds + 16 * 2048;                     // 16*512  f32 (32 KB)
  float* s_sum = c_lds + 16 * 512;                      // 16 f32
  float* s_sq  = s_sum + 16;                            // 16 f32
  unsigned short* h_lds = (unsigned short*)(s_sq + 16); // 16*512 bf16 (16 KB)

  const int tid  = threadIdx.x;
  const int lane = tid & 31;
  const int wave = tid >> 5;           // 0..31
  const int b0   = blockIdx.x * 16;    // this WG owns batch rows [b0, b0+16)

  // init h, c from broadcast init state; zero stats
  for (int q = 0; q < 8; ++q) {
    int p = tid * 8 + q;
    int m = p >> 9, j = p & 511;
    h_lds[m * 512 + j] = f2bf(h0[j]);
    c_lds[m * 512 + j] = c0[j];
  }
  if (tid < 16) { s_sum[tid] = 0.f; s_sq[tid] = 0.f; }
  __syncthreads();

  const int mA   = lane & 15;
  const int asel = (lane < 16) ? 0 : 8;
  const int lcol = lane & 15;
  const int bsel = (lane < 16) ? 0 : 8;

  for (int t = 0; t < 1024; ++t) {
    const float* zx_t = Zx + ((size_t)t * 64 + b0) * 2048;
    if (t + 1 < 1024)   // hide next step's 128 KB Zx read behind the GEMM
      __builtin_prefetch(
          (const char*)(Zx + ((size_t)(t + 1) * 64 + b0) * 2048) + (size_t)tid * 128,
          0, 1);

    // ---- GEMM: Zh tile (+Zx) -> z_lds; each wave does 4 N-tiles ----
    for (int tt2 = 0; tt2 < 4; ++tt2) {
      const int n0 = (wave + 32 * tt2) * 16;
      const unsigned short* brow = Whb + (size_t)(n0 + lcol) * 512;
      v8f acc = {0.f, 0.f, 0.f, 0.f, 0.f, 0.f, 0.f, 0.f};
      for (int k0 = 0; k0 < 512; k0 += 32) {
        BFrag a, b;
        a.h.lo = *(const v8us*)(h_lds + mA * 512 + k0 + asel);       // ds_load_b128
        a.h.hi = *(const v8us*)(h_lds + mA * 512 + k0 + 16 + asel);
        b.h.lo = *(const v8us*)(brow + k0 + bsel);                   // L2-resident
        b.h.hi = *(const v8us*)(brow + k0 + 16 + bsel);
        acc = __builtin_amdgcn_wmma_f32_16x16x32_bf16(
            false, a.v, false, b.v, (short)0, acc, false, false);
      }
#pragma unroll
      for (int r = 0; r < 8; ++r) {
        int m = (lane < 16) ? r : (8 + r);
        int n = n0 + lcol;
        z_lds[m * 2048 + n] = acc[r] + zx_t[m * 2048 + n];
      }
    }
    __syncthreads();

    // ---- layernorm stats: 64 threads per row, 32 elems each ----
    {
      int m = tid >> 6;
      const float* zr = z_lds + m * 2048 + (tid & 63) * 32;
      float s = 0.f, q2 = 0.f;
#pragma unroll
      for (int i = 0; i < 32; ++i) { float v = zr[i]; s += v; q2 += v * v; }
      atomicAdd(&s_sum[m], s);   // ds_add_f32
      atomicAdd(&s_sq[m],  q2);
    }
    __syncthreads();

    // ---- gates + state update + output ----
    for (int q = 0; q < 8; ++q) {
      int p = tid * 8 + q;
      int m = p >> 9, j = p & 511;
      float mu  = s_sum[m] * (1.f / 2048.f);
      float var = s_sq[m] * (1.f / 2048.f) - mu * mu;
      float rs  = rsqrtf(var + 1e-5f);
      const float* zr = z_lds + m * 2048;
      float z0 = (zr[j]        - mu) * rs * gamma[j]        + beta[j];
      float z1 = (zr[512 + j]  - mu) * rs * gamma[512 + j]  + beta[512 + j];
      float z2 = (zr[1024 + j] - mu) * rs * gamma[1024 + j] + beta[1024 + j];
      float z3 = (zr[1536 + j] - mu) * rs * gamma[1536 + j] + beta[1536 + j];
      float fg = 1.f / (1.f + __expf(-z0));
      float ig = 1.f / (1.f + __expf(-z1));
      float og = 1.f / (1.f + __expf(-z2));
      float gl = 0.5f * z3 * (1.f + erff(z3 * 0.70710678118f)); // exact gelu
      float cn = fg * c_lds[m * 512 + j] + ig * gl;
      float hn = og * cn;
      c_lds[m * 512 + j] = cn;
      h_lds[m * 512 + j] = f2bf(hn);
      out[(size_t)(b0 + m) * (1024 * 512) + (size_t)t * 512 + j] = hn;
    }
    __syncthreads();            // gates done: z/h/c safe to overwrite
    if (tid < 16) { s_sum[tid] = 0.f; s_sq[tid] = 0.f; }  // ordered by next GEMM barrier
  }
}

// ---------------- host ----------------
extern "C" void kernel_launch(void* const* d_in, const int* in_sizes, int n_in,
                              void* d_out, int out_size, void* d_ws, size_t ws_size,
                              hipStream_t stream) {
  const float* x     = (const float*)d_in[0];
  const float* W     = (const float*)d_in[1];
  const float* gamma = (const float*)d_in[2];
  const float* beta  = (const float*)d_in[3];
  const float* h0    = (const float*)d_in[4];
  const float* c0    = (const float*)d_in[5];
  float* out = (float*)d_out;

  char* ws = (char*)d_ws;
  unsigned short* Wxb = (unsigned short*)ws;                    //  2 MB
  unsigned short* Whb = (unsigned short*)(ws + (2ull << 20));   //  2 MB
  unsigned short* xb  = (unsigned short*)(ws + (4ull << 20));   // 64 MB
  float*          Zx  = (float*)(ws + (68ull << 20));           // 512 MB

  k_conv_w<<<(2048 * 1024) / 256, 256, 0, stream>>>(W, Wxb, Whb);
  k_conv_x<<<(64 * 1024 * 512) / 256, 256, 0, stream>>>(x, xb);
  k_gemm_zx<<<65536, 256, 0, stream>>>(xb, Wxb, Zx);

  size_t smem = (size_t)(16 * 2048 * 4 + 16 * 512 * 4 + 32 * 4 + 16 * 512 * 2);
  k_lstm<<<4, 1024, smem, stream>>>(Zx, Whb, gamma, beta, h0, c0, out);
}
